// CrossAttention_53326313947704
// MI455X (gfx1250) — compile-verified, matrix operand order checked by reading, not run on hardware
//
#include <hip/hip_runtime.h>
#include <hip/hip_bf16.h>
#include <math.h>

typedef _Float16 v16h __attribute__((ext_vector_type(16)));
typedef _Float16 v8h  __attribute__((ext_vector_type(8)));
typedef float    v8f  __attribute__((ext_vector_type(8)));

#define C_     256
#define C8_    32
#define N_     8192
#define NT_    512        // N/16 tiles
#define KSTEPS 8          // C/32

// Half-index -> K mapping for 16-bit A/B WMMA operands (ISA 7.12.2):
// lanes 0-15: halves 0..7 -> K 0..7, halves 8..15 -> K 16..23
// lanes 16-31: +8
__device__ __forceinline__ int kmap(int i, int hi) {
    return (i < 8 ? i : i + 8) + (hi ? 8 : 0);
}

// ---------------- K1: pack 6 weight matrices into A-operand tiles ----------------
// Tile order: qA[16] kA[16] vA[128] qB[16] kB[16] vB[128]  (tile = cot*8 + kstep)
__global__ void k1_pack_weights(const float* WqA, const float* WkA, const float* WvA,
                                const float* WqB, const float* WkB, const float* WvB,
                                _Float16* wpack) {
    int idx = blockIdx.x * 256 + threadIdx.x;
    if (idx >= 320 * 512) return;
    int tile = idx >> 9, w = idx & 511;
    int lane = w >> 4, i = w & 15, hi = (lane >= 16);
    const float* W; int local;
    if      (tile < 16)  { W = WqA; local = tile;       }
    else if (tile < 32)  { W = WkA; local = tile - 16;  }
    else if (tile < 160) { W = WvA; local = tile - 32;  }
    else if (tile < 176) { W = WqB; local = tile - 160; }
    else if (tile < 192) { W = WkB; local = tile - 176; }
    else                 { W = WvB; local = tile - 192; }
    int cot = local >> 3, kst = local & 7;
    int co = cot * 16 + (lane & 15);
    int c  = kst * 32 + kmap(i, hi);
    wpack[idx] = (_Float16)W[co * C_ + c];
}

// ---------------- K2: pack x (f32 [C,N]) into B-operand tiles [kst][nt][512] -----
__global__ void k2_pack_x(const float* x, _Float16* xpack) {
    int idx = blockIdx.x * 256 + threadIdx.x;
    if (idx >= KSTEPS * NT_ * 512) return;
    int tile = idx >> 9, w = idx & 511;
    int kst = tile / NT_, nt = tile % NT_;
    int lane = w >> 4, i = w & 15, hi = (lane >= 16);
    int n = nt * 16 + (lane & 15);
    int c = kst * 32 + kmap(i, hi);
    xpack[idx] = (_Float16)x[(size_t)c * N_ + n];
}

// ---------------- K3: projection GEMM O = W*X + b via WMMA ----------------
// One wave per 16x16 output tile. Writes row-major f16 (n-major) Q/K/V.
__global__ void __launch_bounds__(256) k3_proj(
        const _Float16* wpack, const _Float16* xpack,
        const float* bqA, const float* bkA, const float* bvA,
        const float* bqB, const float* bkB, const float* bvB,
        _Float16* qrow, _Float16* krow, _Float16* vrow) {
    int wid  = blockIdx.x * 8 + (threadIdx.x >> 5);   // 0..20479
    int lane = threadIdx.x & 31;
    int task = wid / 10240;
    int r    = wid % 10240;
    const float* bias; _Float16* out; int ldo, cot, nt, matBase;
    if (r < 1024) {
        cot = r >> 9; nt = r & 511;
        bias = task ? bqB : bqA; out = qrow + (size_t)task * N_ * C8_;
        ldo = C8_; matBase = task ? 160 : 0;
    } else if (r < 2048) {
        r -= 1024; cot = r >> 9; nt = r & 511;
        bias = task ? bkB : bkA; out = krow + (size_t)task * N_ * C8_;
        ldo = C8_; matBase = task ? 176 : 16;
    } else {
        r -= 2048; cot = r >> 9; nt = r & 511;
        bias = task ? bvB : bvA; out = vrow + (size_t)task * N_ * C_;
        ldo = C_;  matBase = task ? 192 : 32;
    }
    int hi = (lane >= 16);
    v8f acc = {0.f,0.f,0.f,0.f,0.f,0.f,0.f,0.f};
#pragma unroll
    for (int k = 0; k < KSTEPS; ++k) {
        v16h a = *(const v16h*)(wpack + ((size_t)(matBase + cot * 8 + k)) * 512 + lane * 16);
        v16h b = *(const v16h*)(xpack + ((size_t)(k * NT_ + nt)) * 512 + lane * 16);
        acc = __builtin_amdgcn_wmma_f32_16x16x32_f16(false, a, false, b, (short)0, acc, false, false);
    }
    int n = nt * 16 + (lane & 15);
    int cobase = cot * 16 + (hi ? 8 : 0);
    v8h st;
#pragma unroll
    for (int j = 0; j < 8; ++j) st[j] = (_Float16)(acc[j] + bias[cobase + j]);
    *(v8h*)(out + (size_t)n * ldo + cobase) = st;   // contiguous 16B store
}

// ---------------- K4: repack Q/K/V into operand-ready swizzled tiles ----------------
// qpack[t][nt][512]: B-operand of Q[32c x 16n]; kpack[t][mt][512]: A-operand of K^T[16m x 32c]
// vpack[t][mt32*16+ct][512]: B-operand of V^T[32m x 16c]
__global__ void k4_repack(const _Float16* qrow, const _Float16* krow, const _Float16* vrow,
                          _Float16* qpack, _Float16* kpack, _Float16* vpack) {
    int idx = blockIdx.x * 256 + threadIdx.x;
    const int QH = 2 * NT_ * 512, KH = QH, VH = 2 * 256 * 16 * 512;
    if (idx < QH) {
        int t = idx / (NT_ * 512), rem = idx % (NT_ * 512);
        int nt = rem >> 9, w = rem & 511;
        int lane = w >> 4, i = w & 15, hi = (lane >= 16);
        int n = nt * 16 + (lane & 15), c = kmap(i, hi);
        qpack[idx] = qrow[(size_t)t * N_ * C8_ + (size_t)n * C8_ + c];
    } else if (idx < QH + KH) {
        int j = idx - QH;
        int t = j / (NT_ * 512), rem = j % (NT_ * 512);
        int mt = rem >> 9, w = rem & 511;
        int lane = w >> 4, i = w & 15, hi = (lane >= 16);
        int m = mt * 16 + (lane & 15), c = kmap(i, hi);
        kpack[j] = krow[(size_t)t * N_ * C8_ + (size_t)m * C8_ + c];
    } else if (idx < QH + KH + VH) {
        int j = idx - QH - KH;
        int t = j / (256 * 16 * 512), rem = j % (256 * 16 * 512);
        int tile = rem >> 9, w = rem & 511;
        int mt = tile >> 4, ct = tile & 15;
        int lane = w >> 4, i = w & 15, hi = (lane >= 16);
        int c = ct * 16 + (lane & 15);
        int m = mt * 32 + kmap(i, hi);
        vpack[j] = vrow[(size_t)t * N_ * C_ + (size_t)m * C_ + c];
    }
}

// ---------------- K5: fused flash cross-attention + gamma*F + x ----------------
// Workgroup = 8 waves = 8 consecutive n-tiles of one task. Per 32-m chunk the K (2KB)
// and V (16KB) operand tiles are DMA'd ONCE into LDS via global_load_async_to_lds_b128
// (double-buffered, ASYNCcnt + workgroup barrier), then all 8 waves consume them from
// LDS -> 8x less L2 traffic and VMEM pipe left free for the async copy stream.
#define CHUNK_HALVES 9216            // (2 + 16) tiles * 512 halves = 18KB
#define CHUNK_XFERS  1152            // 18KB / 16B

__global__ void __launch_bounds__(256, 1) k5_flash(
        const _Float16* qpack, const _Float16* kpack, const _Float16* vpack,
        const float* x, const float* gamma_A, const float* gamma_B, float* out) {
    __shared__ _Float16 smem[2 * CHUNK_HALVES];   // 36KB double buffer

    int tid  = threadIdx.x;
    int widx = tid >> 5;
    int lane = tid & 31;
    int task = blockIdx.x >> 6;                   // 64 blocks per task
    int nt   = (blockIdx.x & 63) * 8 + widx;      // this wave's n-tile
    int hi   = (lane >= 16);
    // task 0: out1 = gA*F_A + x (Q=QA, K=KB, V=VB); task 1: out2 (Q=QB, K=KA, V=VA)
    int qsel = task, kvsel = 1 - task;
    const _Float16* qb = qpack + (size_t)qsel  * NT_ * 512;
    const _Float16* kb = kpack + (size_t)kvsel * NT_ * 512;
    const _Float16* vb = vpack + (size_t)kvsel * 256 * 16 * 512;
    float gamma = task ? gamma_B[0] : gamma_A[0];

    v16h bq = *(const v16h*)(qb + (size_t)nt * 512 + lane * 16);
    const v8f z8 = {0.f,0.f,0.f,0.f,0.f,0.f,0.f,0.f};

    // ---- pass A: column max over all m (direct global loads, K only) ----
    float vmax = -3.0e38f;
    for (int mt = 0; mt < NT_; ++mt) {
        v16h a = *(const v16h*)(kb + (size_t)mt * 512 + lane * 16);
        v8f s = __builtin_amdgcn_wmma_f32_16x16x32_f16(false, a, false, bq, (short)0, z8, false, false);
#pragma unroll
        for (int j = 0; j < 8; ++j) vmax = fmaxf(vmax, s[j]);
    }
    vmax = fmaxf(vmax, __shfl_xor(vmax, 16, 32));

    // ---- pass B: exp + PV accumulate, LDS-staged operands ----
    v8f acc[16];
#pragma unroll
    for (int ct = 0; ct < 16; ++ct) acc[ct] = z8;
    float lsum = 0.f;

    const char* kbase = (const char*)kb;   // chunk mc: 2KB at mc*2048
    const char* vbase = (const char*)vb;   // chunk mc: 16KB at mc*16384

    // stage chunk 0 into buffer 0
    {
        unsigned ldsb = (unsigned)(size_t)(&smem[0]);
        for (int u = tid; u < CHUNK_XFERS; u += 256) {
            unsigned lds = ldsb + u * 16;
            const char* g = (u < 128) ? (kbase + u * 16) : (vbase + (u - 128) * 16);
            unsigned long long ga = (unsigned long long)(size_t)g;
            asm volatile("global_load_async_to_lds_b128 %0, %1, off"
                         :: "v"(lds), "v"(ga) : "memory");
        }
    }
    asm volatile("s_wait_asynccnt 0x0" ::: "memory");
    __syncthreads();

    for (int mc = 0; mc < NT_ / 2; ++mc) {        // 256 chunks of 32 m
        // prefetch next chunk into the other buffer
        if (mc + 1 < NT_ / 2) {
            unsigned ldsb = (unsigned)(size_t)(&smem[((mc + 1) & 1) * CHUNK_HALVES]);
            const char* kc = kbase + (size_t)(mc + 1) * 2048;
            const char* vc = vbase + (size_t)(mc + 1) * 16384;
            for (int u = tid; u < CHUNK_XFERS; u += 256) {
                unsigned lds = ldsb + u * 16;
                const char* g = (u < 128) ? (kc + u * 16) : (vc + (u - 128) * 16);
                unsigned long long ga = (unsigned long long)(size_t)g;
                asm volatile("global_load_async_to_lds_b128 %0, %1, off"
                             :: "v"(lds), "v"(ga) : "memory");
            }
        }
        // consume current buffer from LDS
        const _Float16* buf = &smem[(mc & 1) * CHUNK_HALVES];
        v16h a0 = *(const v16h*)(buf + lane * 16);
        v16h a1 = *(const v16h*)(buf + 512 + lane * 16);
        v8f s0 = __builtin_amdgcn_wmma_f32_16x16x32_f16(false, a0, false, bq, (short)0, z8, false, false);
        v8f s1 = __builtin_amdgcn_wmma_f32_16x16x32_f16(false, a1, false, bq, (short)0, z8, false, false);
        v16h ap;                                   // P^T A-operand: pure per-lane pack
#pragma unroll
        for (int j = 0; j < 8; ++j) {
            float e0 = __expf(s0[j] - vmax);
            float e1 = __expf(s1[j] - vmax);
            ap[j]     = (_Float16)e0;
            ap[j + 8] = (_Float16)e1;
            lsum += e0 + e1;
        }
#pragma unroll
        for (int ct = 0; ct < 16; ++ct) {
            v16h bv = *(const v16h*)(buf + 1024 + ct * 512 + lane * 16);
            acc[ct] = __builtin_amdgcn_wmma_f32_16x16x32_f16(false, ap, false, bv, (short)0, acc[ct], false, false);
        }
        // all of this wave's async transfers done + all waves finished reading
        asm volatile("s_wait_asynccnt 0x0" ::: "memory");
        __syncthreads();
    }

    // row (n) normalizers for this lane's accumulator rows
    float linv[8];
#pragma unroll
    for (int r = 0; r < 8; ++r) {
        int src = r + (hi ? 8 : 0);
        float lt = __shfl(lsum, src, 32) + __shfl(lsum, src + 16, 32);
        linv[r] = 1.0f / lt;
    }
    // store: acc D-layout has lane<->c column, VGPR<->n row
    int col = lane & 15;
    size_t obase = (size_t)task * C_ * N_;
#pragma unroll
    for (int ct = 0; ct < 16; ++ct) {
        int c = ct * 16 + col;
#pragma unroll
        for (int r = 0; r < 8; ++r) {
            int n = nt * 16 + r + (hi ? 8 : 0);
            out[obase + (size_t)c * N_ + n] = gamma * acc[ct][r] * linv[r] + x[(size_t)c * N_ + n];
        }
    }
}

// ---------------- host launcher ----------------
extern "C" void kernel_launch(void* const* d_in, const int* in_sizes, int n_in,
                              void* d_out, int out_size, void* d_ws, size_t ws_size,
                              hipStream_t stream) {
    (void)in_sizes; (void)n_in; (void)out_size; (void)ws_size;
    const float* x   = (const float*)d_in[0];
    const float* WqA = (const float*)d_in[1];  const float* bqA = (const float*)d_in[2];
    const float* WkA = (const float*)d_in[3];  const float* bkA = (const float*)d_in[4];
    const float* WvA = (const float*)d_in[5];  const float* bvA = (const float*)d_in[6];
    const float* WqB = (const float*)d_in[7];  const float* bqB = (const float*)d_in[8];
    const float* WkB = (const float*)d_in[9];  const float* bkB = (const float*)d_in[10];
    const float* WvB = (const float*)d_in[11]; const float* bvB = (const float*)d_in[12];
    const float* gA  = (const float*)d_in[13]; const float* gB  = (const float*)d_in[14];
    float* out = (float*)d_out;

    char* ws = (char*)d_ws;
    size_t off = 0;
    auto carve = [&](size_t bytes) -> _Float16* {
        _Float16* p = (_Float16*)(ws + off);
        off += (bytes + 255) & ~(size_t)255;
        return p;
    };
    _Float16* xpack = carve((size_t)KSTEPS * NT_ * 512 * 2);  // 4 MiB
    _Float16* wpack = carve((size_t)320 * 512 * 2);           // 320 KiB
    _Float16* qrow  = carve((size_t)2 * N_ * C8_ * 2);        // 1 MiB
    _Float16* krow  = carve((size_t)2 * N_ * C8_ * 2);        // 1 MiB
    _Float16* vrow  = carve((size_t)2 * N_ * C_ * 2);         // 8 MiB
    _Float16* qpack = carve((size_t)2 * NT_ * 512 * 2);       // 1 MiB
    _Float16* kpack = carve((size_t)2 * NT_ * 512 * 2);       // 1 MiB
    _Float16* vpack = carve((size_t)2 * 256 * 16 * 512 * 2);  // 8 MiB

    k1_pack_weights<<<(320 * 512 + 255) / 256, 256, 0, stream>>>(WqA, WkA, WvA, WqB, WkB, WvB, wpack);
    k2_pack_x<<<(KSTEPS * NT_ * 512 + 255) / 256, 256, 0, stream>>>(x, xpack);
    k3_proj<<<2560, 256, 0, stream>>>(wpack, xpack, bqA, bkA, bvA, bqB, bkB, bvB, qrow, krow, vrow);
    int repack_threads = 2 * NT_ * 512 * 2 + 2 * 256 * 16 * 512;
    k4_repack<<<(repack_threads + 255) / 256, 256, 0, stream>>>(qrow, krow, vrow, qpack, kpack, vpack);
    k5_flash<<<128, 256, 0, stream>>>(qpack, kpack, vpack, x, gA, gB, out);
}
